// EdgeFlipping_83915071030249
// MI455X (gfx1250) — matrix operand order repaired and checked by reading, not compiled, not run on hardware
//
#include <hip/hip_runtime.h>

typedef unsigned short ushort_t;
typedef __attribute__((ext_vector_type(16))) __bf16 v16bf;
typedef __attribute__((ext_vector_type(8)))  float  v8f;

#define NDIM 4096
#define EPSF 1e-10f
#define BS_STRIDE 132   // 128 + 4 pad: 264B rows -> 66 words, 66*16 % 64 = 32 (conflict-free)
#define AS_STRIDE 36    // 32 + 4 pad: 72B rows, 8B-aligned chunks

// ---------- helpers ----------
__device__ __forceinline__ float bf2f(unsigned h) {
    return __uint_as_float(h << 16);
}
__device__ __forceinline__ ushort_t f2bf(float f) {  // round-to-nearest-even
    unsigned u = __float_as_uint(f);
    unsigned r = u + 0x7FFFu + ((u >> 16) & 1u);
    return (ushort_t)(r >> 16);
}

// A-matrix fragment (contiguous K layout): row base in LDS, klo = (lane>=16)?16:0.
// elements 0..15 -> K=klo..klo+15
__device__ __forceinline__ v16bf frag_a(const ushort_t* row, int klo) {
    union { v16bf v; unsigned long long q[4]; } f;
    const unsigned long long* s0 = (const unsigned long long*)(row + klo);
    f.q[0] = s0[0]; f.q[1] = s0[1]; f.q[2] = s0[2]; f.q[3] = s0[3];
    return f.v;
}
// B-matrix fragment from natural [k][j] LDS tile: n = j (lane_lo), K = kbase + 0..15
__device__ __forceinline__ v16bf frag_b(const ushort_t* Bs, int kbase, int j) {
    union { v16bf v; ushort_t u[16]; } f;
    #pragma unroll
    for (int t = 0; t < 16; ++t) f.u[t] = Bs[(kbase + t) * BS_STRIDE + j];
    return f.v;
}
// |x| on a packed bf16 fragment: clear sign bits
__device__ __forceinline__ v16bf frag_abs(v16bf x) {
    union { v16bf v; unsigned long long q[4]; } f; f.v = x;
    #pragma unroll
    for (int t = 0; t < 4; ++t) f.q[t] &= 0x7FFF7FFF7FFF7FFFull;
    return f.v;
}

// ---------- kernel 1: mod_adj (elementwise, straight-through forward value) ----------
__global__ void ef_modadj(const float* __restrict__ ori,
                          const float* __restrict__ cp,
                          const float* __restrict__ u,
                          float* __restrict__ out) {
    const size_t gid = (size_t)blockIdx.x * blockDim.x + threadIdx.x;
    const float4 o4 = ((const float4*)ori)[gid];
    const float4 p4 = ((const float4*)cp)[gid];
    const float4 u4 = ((const float4*)u)[gid];
    float4 r;
    #pragma unroll
    for (int c = 0; c < 4; ++c) {
        float o  = (&o4.x)[c];
        float p  = fminf(fmaxf((&p4.x)[c], EPSF), 1.0f - EPSF);
        float uu = (&u4.x)[c];
        float x  = (__logf(p) - log1pf(-p)) + (__logf(uu) - log1pf(-uu));
        // hard = round(sigmoid(x/tau)) = (x > 0);  mod = ori*(1-2*hard)
        (&r.x)[c] = (x > 0.0f) ? -o : o;
    }
    ((float4*)out)[gid] = r;
}

// ---------- kernel 2: A = M + M^T - diag(M) broadcast; emit bf16 A ----------
__global__ void ef_buildA(const float* __restrict__ M,
                          ushort_t* __restrict__ Abf) {
    __shared__ float tA[32][33];
    __shared__ float tB[32][33];
    __shared__ float dg[32];
    const int tx = threadIdx.x, ty = threadIdx.y;
    const int I0 = blockIdx.y * 32, J0 = blockIdx.x * 32;
    #pragma unroll
    for (int k = 0; k < 4; ++k) {
        int iy = ty + k * 8;
        tA[iy][tx] = M[(size_t)(I0 + iy) * NDIM + J0 + tx];
        tB[iy][tx] = M[(size_t)(J0 + iy) * NDIM + I0 + tx];
    }
    if (ty == 0) dg[tx] = M[(size_t)(J0 + tx) * NDIM + (J0 + tx)];
    __syncthreads();
    #pragma unroll
    for (int k = 0; k < 4; ++k) {
        int iy = ty + k * 8;
        float a = tA[iy][tx] + tB[tx][iy] - dg[tx];
        Abf[(size_t)(I0 + iy) * NDIM + J0 + tx] = f2bf(a);
    }
}

// ---------- kernel 3: fused tr(G^3) and tr(|G|^3) via WMMA ----------
// grid (32,32), block 512 = 16 waves; each block does a 128x128 tile of (G@G)
// and of (|G|@|G|), dotted against G^T / |G^T| without materializing either product.
// Wave grid: 2 m-groups x 8 n -> each wave: 4 m-subtiles, ONE B fragment
// (B-fragment gathers are the expensive LDS path; A-fragments are fused b64 pairs).
__global__ void __launch_bounds__(512)
ef_trace2(const ushort_t* __restrict__ G, float* __restrict__ accum /* [0]=signed [1]=abs */) {
    __shared__ ushort_t As[128 * AS_STRIDE];  // [i][k]
    __shared__ ushort_t Bs[32 * BS_STRIDE];   // [k][j] natural layout
    __shared__ float    redS[512];
    __shared__ float    redA[512];

    const int tid  = threadIdx.x;
    const int lane = tid & 31;
    const int wave = tid >> 5;                 // 0..15
    const int wm = wave >> 3, wn = wave & 7;   // wm: 0..1 (4 msubs each), wn: 0..7 (1 nsub)
    const int lane_hi = lane >> 4, lane_lo = lane & 15;
    const int I0 = blockIdx.y * 128, J0 = blockIdx.x * 128;

    // staging coordinates (loop-invariant)
    const int sa_i = tid >> 2, sa_seg = tid & 3;    // As: 128 rows x 4 chunks
    const int sb_k = tid >> 4, sb_seg = tid & 15;   // Bs: 32 rows x 16 chunks

    v8f accS[4], accA[4];
    #pragma unroll
    for (int s = 0; s < 4; ++s) { accS[s] = (v8f){}; accA[s] = (v8f){}; }

    for (int k0 = 0; k0 < NDIM; k0 += 32) {
        // stage As: rows I0..I0+127 x cols k0..k0+31 ; one b128 load + two b64 stores/thread
        {
            const uint4 q = *(const uint4*)(G + (size_t)(I0 + sa_i) * NDIM + k0 + sa_seg * 8);
            unsigned long long lo = ((unsigned long long)q.y << 32) | q.x;
            unsigned long long hi = ((unsigned long long)q.w << 32) | q.z;
            *(unsigned long long*)(As + sa_i * AS_STRIDE + sa_seg * 8)     = lo;
            *(unsigned long long*)(As + sa_i * AS_STRIDE + sa_seg * 8 + 4) = hi;
        }
        // stage Bs (natural): rows k0..k0+31 x cols J0..J0+127
        {
            const uint4 q = *(const uint4*)(G + (size_t)(k0 + sb_k) * NDIM + J0 + sb_seg * 8);
            unsigned long long lo = ((unsigned long long)q.y << 32) | q.x;
            unsigned long long hi = ((unsigned long long)q.w << 32) | q.z;
            *(unsigned long long*)(Bs + sb_k * BS_STRIDE + sb_seg * 8)     = lo;
            *(unsigned long long*)(Bs + sb_k * BS_STRIDE + sb_seg * 8 + 4) = hi;
        }
        // prefetch next K-slab into caches (global_prefetch_b8; no LOADcnt impact)
        if (k0 + 32 < NDIM) {
            __builtin_prefetch(G + (size_t)(I0 + sa_i) * NDIM + (k0 + 32) + sa_seg * 8, 0, 3);
            __builtin_prefetch(G + (size_t)(k0 + 32 + sb_k) * NDIM + J0 + sb_seg * 8, 0, 3);
        }
        __syncthreads();

        v16bf b  = frag_b(Bs, lane_hi * 16, wn * 16 + lane_lo);
        v16bf ba = frag_abs(b);
        #pragma unroll
        for (int mi = 0; mi < 4; ++mi) {
            int msub = wm * 4 + mi;
            v16bf af = frag_a(As + (msub * 16 + lane_lo) * AS_STRIDE, lane_hi * 16);
            v16bf aa = frag_abs(af);
            accS[mi] = __builtin_amdgcn_wmma_f32_16x16x32_bf16(
                false, af, false, b,  (short)0, accS[mi], false, false);
            accA[mi] = __builtin_amdgcn_wmma_f32_16x16x32_bf16(
                false, aa, false, ba, (short)0, accA[mi], false, false);
        }
        __syncthreads();
    }

    // partial traces: D layout -> lane n = lane_lo ; vgpr r -> m = r + lane_hi*8
    float s = 0.0f, sa = 0.0f;
    #pragma unroll
    for (int mi = 0; mi < 4; ++mi) {
        int msub = wm * 4 + mi, nsub = wn;
        int ig = I0 + msub * 16 + lane_hi * 8;      // 8 consecutive i
        int jg = J0 + nsub * 16 + lane_lo;
        const uint4 q = *(const uint4*)(G + (size_t)jg * NDIM + ig);
        const unsigned w[4] = {q.x, q.y, q.z, q.w};
        v8f dS = accS[mi];
        v8f dA = accA[mi];
        #pragma unroll
        for (int t = 0; t < 4; ++t) {
            float v0 = bf2f(w[t] & 0xffffu);
            float v1 = bf2f(w[t] >> 16);
            s  += dS[2 * t] * v0 + dS[2 * t + 1] * v1;
            sa += dA[2 * t] * fabsf(v0) + dA[2 * t + 1] * fabsf(v1);
        }
    }
    redS[tid] = s;
    redA[tid] = sa;
    __syncthreads();
    #pragma unroll
    for (int off = 256; off > 0; off >>= 1) {
        if (tid < off) { redS[tid] += redS[tid + off]; redA[tid] += redA[tid + off]; }
        __syncthreads();
    }
    if (tid == 0) {
        atomicAdd(&accum[0], redS[0]);
        atomicAdd(&accum[1], redA[0]);
    }
}

// ---------- tiny kernels ----------
__global__ void ef_zero(float* acc) { acc[0] = 0.0f; acc[1] = 0.0f; }

__global__ void ef_balance(const float* acc, float* out) {
    out[(size_t)NDIM * NDIM] = 0.5f * (1.0f + acc[0] / acc[1]);
}

// ---------- launcher ----------
extern "C" void kernel_launch(void* const* d_in, const int* in_sizes, int n_in,
                              void* d_out, int out_size, void* d_ws, size_t ws_size,
                              hipStream_t stream) {
    const float* ori = (const float*)d_in[0];
    const float* cp  = (const float*)d_in[1];
    const float* u   = (const float*)d_in[2];
    float* out = (float*)d_out;

    char* ws = (char*)d_ws;
    ushort_t* Abf = (ushort_t*)ws;                              // 33.5 MB (L2-resident)
    float*    acc = (float*)(ws + (size_t)NDIM * NDIM * 2);     // 8 B

    // 1) mod_adj -> d_out (float4 per thread)
    const size_t nElem = (size_t)NDIM * NDIM;
    ef_modadj<<<dim3((unsigned)(nElem / 4 / 256)), dim3(256), 0, stream>>>(ori, cp, u, out);

    // 2) A as bf16 in workspace (|A| derived in-register later)
    ef_buildA<<<dim3(NDIM / 32, NDIM / 32), dim3(32, 8), 0, stream>>>(out, Abf);

    // 3) fused signed+abs trace via WMMA
    ef_zero<<<1, 1, 0, stream>>>(acc);
    ef_trace2<<<dim3(NDIM / 128, NDIM / 128), dim3(512), 0, stream>>>(Abf, acc);

    // 4) balance scalar
    ef_balance<<<1, 1, 0, stream>>>(acc, out);
}
